// MODEL_89507118449162
// MI455X (gfx1250) — matrix operand order repaired and checked by reading, not compiled, not run on hardware
//
#include <hip/hip_runtime.h>
#include <hip/hip_bf16.h>
#include <math.h>

// ---------------- problem constants ----------------
#define BSZ   8
#define TT    64
#define NN    200
#define DD    128
#define CC1   101      // C+1
#define HH    256
#define FDIM  231      // 101 + 128 + 2
#define GG    (BSZ*TT)         // 512 graphs
#define MROWS (GG*NN)          // 102400 node rows
#define LKP   256              // LSTM input K padded (231 -> 256)

typedef __attribute__((ext_vector_type(16))) __bf16 v16bf;
typedef __attribute__((ext_vector_type(8)))  float  v8f;

union FragU { uint4 q[2]; v16bf v; };

__device__ __forceinline__ float sigf(float x){ return 1.f/(1.f+__expf(-x)); }

// A fragment: 16x32 bf16 tile, rows rowM..rowM+15, k chunk kc (32 wide), row-major A, lda elems.
// Works for global or LDS pointers (addrspace inferred after inlining).
__device__ __forceinline__ v16bf load_afrag(const __bf16* A, int lda, int rowM, int kc) {
  int l = threadIdx.x & 31;
  int m = rowM + (l & 15);
  int kh = l >> 4;                       // 0 or 1
  const __bf16* p = A + (size_t)m * lda + kc*32 + kh*8;
  FragU f;
  f.q[0] = *(const uint4*)p;             // K = kc*32 + kh*8 + [0..7]
  f.q[1] = *(const uint4*)(p + 16);      // K = kc*32 + 16 + kh*8 + [0..7]
  return f.v;
}

// B fragment from pre-packed layout: ((nt*kchunks + kc)*32 + lane)*16 + j
__device__ __forceinline__ v16bf load_bfrag(const __bf16* P, int kchunks, int nt, int kc) {
  int l = threadIdx.x & 31;
  const __bf16* p = P + (((size_t)nt * kchunks + kc)*32 + l)*16;
  FragU f;
  f.q[0] = *(const uint4*)p;
  f.q[1] = *(const uint4*)(p + 8);
  return f.v;
}

// ---------------- prep kernels ----------------
__global__ void k_ew_table(const float* EEW, float* tab) {
  int t = threadIdx.x;
  if (t < 8) { float s = 0.f; for (int d = 0; d < DD; ++d) s += EEW[t*DD + d]; tab[t] = s / (float)DD; }
}

__global__ void k_ew_gather(const int* etype, const float* tab, float* ew) {
  int i = blockIdx.x*blockDim.x + threadIdx.x;
  if (i < MROWS) ew[i] = tab[etype[i]];
}

__global__ void k_embed(const int* node_id, const float* EW, float* X, __bf16* Xbf) {
  size_t total = (size_t)MROWS * DD;
  for (size_t idx = (size_t)blockIdx.x*blockDim.x + threadIdx.x; idx < total;
       idx += (size_t)gridDim.x*blockDim.x) {
    size_t nodeflat = idx >> 7; int d = (int)(idx & 127);
    int nid = node_id[nodeflat];
    float v = EW[(size_t)nid*DD + d];
    X[idx] = v; Xbf[idx] = (__bf16)v;
  }
}

// Pack B (effective Kp x N, zero-padded beyond Korig) into WMMA B-fragment order.
// trans=0: B[k*N+n] (k<Korig); trans=1: B[n*Korig+k] (k<Korig).
__global__ void k_pack_b(const float* B, __bf16* out, int Kp, int N, int Korig, int trans) {
  int total = Kp*N;
  int kchunks = Kp >> 5;
  for (int idx = blockIdx.x*blockDim.x + threadIdx.x; idx < total; idx += gridDim.x*blockDim.x) {
    int j = idx & 15, lane = (idx >> 4) & 31, rest = idx >> 9;
    int kc = rest % kchunks, nt = rest / kchunks;
    int k = kc*32 + (j>>3)*16 + (lane>>4)*8 + (j&7);
    int n = nt*16 + (lane & 15);
    float v = 0.f;
    if (k < Korig) v = trans ? B[(size_t)n*Korig + k] : B[(size_t)k*N + n];
    out[idx] = (__bf16)v;
  }
}

// ---------------- GGNN layer kernels ----------------
// m = x_bf @ W_l   (M=102400, K=128, N=128), bf16 out. grid (M/64, 2), 256 thr.
__global__ void k_gemm_msg(const __bf16* A, const __bf16* Bp, __bf16* Cbf) {
  int w = threadIdx.x >> 5, l = threadIdx.x & 31;
  int rowM = blockIdx.x*64 + (w & 3)*16;
  int colB = blockIdx.y*64 + (w >> 2)*32;
  int nt0 = colB >> 4;
  v8f z = {0.f,0.f,0.f,0.f,0.f,0.f,0.f,0.f};
  v8f acc0 = z, acc1 = z;
  for (int kc = 0; kc < 4; ++kc) {
    v16bf a  = load_afrag(A, DD, rowM, kc);
    v16bf b0 = load_bfrag(Bp, 4, nt0,   kc);
    v16bf b1 = load_bfrag(Bp, 4, nt0+1, kc);
    acc0 = __builtin_amdgcn_wmma_f32_16x16x32_bf16(false, a, false, b0, (short)0, acc0, false, false);
    acc1 = __builtin_amdgcn_wmma_f32_16x16x32_bf16(false, a, false, b1, (short)0, acc1, false, false);
  }
  int n = l & 15, mb = 8*(l>>4);
  for (int r = 0; r < 8; ++r) {
    Cbf[(size_t)(rowM+mb+r)*DD + colB + n]      = (__bf16)acc0[r];
    Cbf[(size_t)(rowM+mb+r)*DD + colB + 16 + n] = (__bf16)acc1[r];
  }
}

// gather msg by src, scale by ew, scatter-add to dst; in-place aggr over Mbuf.
// One block per graph. Dynamic LDS = 200*64*4 bytes, two D-halves.
__global__ void k_scatter(__bf16* Mbuf, const float* ew, const int* edge) {
  extern __shared__ float aggr_s[];
  int g = blockIdx.x;
  const int* src = edge + (size_t)g*2*NN;
  const int* dst = src + NN;
  for (int half = 0; half < 2; ++half) {
    int dbase = half*64;
    for (int i = threadIdx.x; i < NN*64; i += blockDim.x) aggr_s[i] = 0.f;
    __syncthreads();
    for (int i = threadIdx.x; i < NN*64; i += blockDim.x) {
      int e = i >> 6, d = i & 63;
      int s = src[e], dd = dst[e];
      float v = (float)Mbuf[((size_t)g*NN + s)*DD + dbase + d] * ew[(size_t)g*NN + e];
      atomicAdd(&aggr_s[dd*64 + d], v);             // ds_add_f32
    }
    __syncthreads();
    for (int i = threadIdx.x; i < NN*64; i += blockDim.x) {
      int nno = i >> 6, d = i & 63;
      Mbuf[((size_t)g*NN + nno)*DD + dbase + d] = (__bf16)aggr_s[i];
    }
    __syncthreads();
  }
}

// fused: gi = aggr@Wih^T + bih ; gh = x@Whh^T + bhh ; GRU update -> x, x_bf16.
// One block per 16 rows (grid = MROWS/16). 8 waves * 6 WMMA tiles = 48 (24 gi + 24 gh).
__global__ void k_gru_fused(const __bf16* Aggr, const __bf16* Xbf,
                            const __bf16* PWih, const __bf16* PWhh,
                            const float* bih, const float* bhh,
                            float* X, __bf16* XbfOut) {
  __shared__ float gbuf[2][16][384];
  int w = threadIdx.x >> 5, l = threadIdx.x & 31;
  int rowM = blockIdx.x * 16;
  v8f z = {0.f,0.f,0.f,0.f,0.f,0.f,0.f,0.f};
  v8f acc[6];
  for (int i = 0; i < 6; ++i) acc[i] = z;
  for (int kc = 0; kc < 4; ++kc) {
    v16bf aA = load_afrag(Aggr, DD, rowM, kc);
    v16bf aX = load_afrag(Xbf,  DD, rowM, kc);
    for (int i = 0; i < 6; ++i) {
      int t = w*6 + i;
      bool isGi = t < 24;
      int nt = isGi ? t : (t - 24);
      v16bf b = load_bfrag(isGi ? PWih : PWhh, 4, nt, kc);
      acc[i] = __builtin_amdgcn_wmma_f32_16x16x32_bf16(false, isGi ? aA : aX, false, b,
                                                       (short)0, acc[i], false, false);
    }
  }
  int n = l & 15, mb = 8*(l>>4);
  for (int i = 0; i < 6; ++i) {
    int t = w*6 + i; bool isGi = t < 24; int nt = isGi ? t : (t - 24);
    float bv = (isGi ? bih : bhh)[nt*16 + n];
    for (int r = 0; r < 8; ++r)
      gbuf[isGi ? 0 : 1][mb + r][nt*16 + n] = acc[i][r] + bv;
  }
  __syncthreads();
  for (int idx = threadIdx.x; idx < 16*DD; idx += blockDim.x) {
    int row = idx >> 7, c = idx & 127;
    float i_r = gbuf[0][row][c], i_z = gbuf[0][row][c+128], i_n = gbuf[0][row][c+256];
    float h_r = gbuf[1][row][c], h_z = gbuf[1][row][c+128], h_n = gbuf[1][row][c+256];
    float r  = sigf(i_r + h_r);
    float zz = sigf(i_z + h_z);
    float nn = tanhf(i_n + r*h_n);
    size_t gi_ = (size_t)(rowM + row)*DD + c;
    float xo = X[gi_];
    float xn = (1.f - zz)*nn + zz*xo;
    X[gi_] = xn;
    XbfOut[gi_] = (__bf16)xn;
  }
}

// ---------------- attention pool ----------------
__global__ void k_attn_pool(const float* X, const float* gw, const float* gb, float* codevec) {
  __shared__ float gsc[NN];
  __shared__ float red;
  int g = blockIdx.x, t = threadIdx.x;
  if (t < NN) {
    const float* xr = X + ((size_t)g*NN + t)*DD;
    float acc = 0.f;
    for (int d = 0; d < DD; ++d) acc += xr[d]*gw[d];
    gsc[t] = sigf(acc + gb[0]);
  }
  __syncthreads();
  if (t == 0) { float m = -1e30f; for (int i = 0; i < NN; ++i) m = fmaxf(m, gsc[i]); red = m; }
  __syncthreads();
  if (t < NN) gsc[t] = __expf(gsc[t] - red);
  __syncthreads();
  if (t == 0) { float s = 0.f; for (int i = 0; i < NN; ++i) s += gsc[i]; red = s; }
  __syncthreads();
  if (t < NN) gsc[t] /= red;
  __syncthreads();
  if (t < DD) {
    float acc = 0.f;
    for (int nno = 0; nno < NN; ++nno) acc += gsc[nno]*X[((size_t)g*NN + nno)*DD + t];
    codevec[(size_t)g*DD + t] = acc;
  }
}

// ---------------- LSTM: sequential over T, WMMA gate projections ----------------
// Single block, 1024 threads = 32 waves. Per step: gates[16pad x 1024] =
// in_bf[16x256] @ PWih (K=256, Wih zero-padded 231->256) + h_bf[16x256] @ PWhh.
// Each wave owns 2 N-tiles -> 32 WMMAs/wave/step.
__global__ void k_lstm(const float* c_embed, const float* codevec, const float* cur_result,
                       const __bf16* PWih, const __bf16* PWhh,
                       const float* bih, const float* bhh,
                       float* lstm_out) {
  __shared__ __bf16 in_bf[16][LKP];   // rows 0..7 live, 8..15 zero; k 231..255 zero
  __shared__ __bf16 h_bf[16][HH];     // rows 0..7 live, 8..15 zero
  __shared__ float  c_s[BSZ][HH];
  __shared__ float  gs[BSZ][4*HH];
  int t = threadIdx.x;
  int w = t >> 5, l = t & 31;
  for (int i = t; i < 16*LKP; i += 1024) in_bf[i>>8][i&255] = (__bf16)0.f;
  for (int i = t; i < 16*HH;  i += 1024) h_bf[i>>8][i&255]  = (__bf16)0.f;
  for (int i = t; i < BSZ*HH; i += 1024) c_s[i>>8][i&255]   = 0.f;
  __syncthreads();
  const int nt0 = 2*w;                 // this wave's two N-tiles of gates
  for (int step = 0; step < TT; ++step) {
    // stage this step's inputs (bf16) into LDS
    for (int i = t; i < BSZ*FDIM; i += 1024) {
      int b = i / FDIM, k = i % FDIM;
      size_t g = (size_t)b*TT + step;
      float v;
      if (k < CC1)           v = c_embed[g*CC1 + k];
      else if (k < CC1 + DD) v = codevec[g*DD + (k - CC1)];
      else                   v = cur_result[g*2 + (k - CC1 - DD)];
      in_bf[b][k] = (__bf16)v;
      if (step + 1 < TT && k < CC1)
        __builtin_prefetch(&c_embed[g*CC1 + CC1 + k], 0, 3);  // next step's row
    }
    __syncthreads();
    v8f zv = {0.f,0.f,0.f,0.f,0.f,0.f,0.f,0.f};
    v8f acc0 = zv, acc1 = zv;
    for (int kc = 0; kc < LKP/32; ++kc) {          // input projection
      v16bf a  = load_afrag(&in_bf[0][0], LKP, 0, kc);
      v16bf b0 = load_bfrag(PWih, LKP/32, nt0,   kc);
      v16bf b1 = load_bfrag(PWih, LKP/32, nt0+1, kc);
      acc0 = __builtin_amdgcn_wmma_f32_16x16x32_bf16(false, a, false, b0, (short)0, acc0, false, false);
      acc1 = __builtin_amdgcn_wmma_f32_16x16x32_bf16(false, a, false, b1, (short)0, acc1, false, false);
    }
    for (int kc = 0; kc < HH/32; ++kc) {           // hidden projection
      v16bf a  = load_afrag(&h_bf[0][0], HH, 0, kc);
      v16bf b0 = load_bfrag(PWhh, HH/32, nt0,   kc);
      v16bf b1 = load_bfrag(PWhh, HH/32, nt0+1, kc);
      acc0 = __builtin_amdgcn_wmma_f32_16x16x32_bf16(false, a, false, b0, (short)0, acc0, false, false);
      acc1 = __builtin_amdgcn_wmma_f32_16x16x32_bf16(false, a, false, b1, (short)0, acc1, false, false);
    }
    // lanes 0..15 hold batch rows 0..7; add bias, spill gates to LDS
    {
      int n  = nt0*16 + (l & 15);
      int mb = 8*(l >> 4);
      if (mb == 0) {
        float bv0 = bih[n]      + bhh[n];
        float bv1 = bih[n + 16] + bhh[n + 16];
        for (int r = 0; r < 8; ++r) {
          gs[r][n]      = acc0[r] + bv0;
          gs[r][n + 16] = acc1[r] + bv1;
        }
      }
    }
    __syncthreads();
    // elementwise LSTM cell update
    for (int i = t; i < BSZ*HH; i += 1024) {
      int b = i >> 8, u = i & 255;
      float ig = gs[b][u], fg = gs[b][u+HH], gg = gs[b][u+2*HH], og = gs[b][u+3*HH];
      float cc = sigf(fg)*c_s[b][u] + sigf(ig)*tanhf(gg);
      float hh = sigf(og)*tanhf(cc);
      c_s[b][u] = cc;
      h_bf[b][u] = (__bf16)hh;
      lstm_out[((size_t)b*TT + step)*HH + u] = hh;
    }
    __syncthreads();
  }
}

// ---------------- prediction + masked BCE ----------------
__global__ void k_init_red(float* red) { if (threadIdx.x < 2) red[threadIdx.x] = 0.f; }

__global__ void k_predict(const float* lstm_out, const float* predW, const float* predb,
                          const float* target_c, const float* result,
                          float* out, float* red) {
  __shared__ float sred[256];
  int g = blockIdx.x, t = threadIdx.x;
  float wv = 0.f, bsum = 0.f, ncnt = 0.f;
  for (int c = 0; c < CC1; ++c) {
    float tc = target_c[(size_t)g*CC1 + c];
    wv   += tc * predW[(size_t)c*HH + t];
    bsum += tc * predb[c];
    ncnt += tc;
  }
  sred[t] = lstm_out[(size_t)g*HH + t] * wv;
  __syncthreads();
  for (int s = 128; s > 0; s >>= 1) { if (t < s) sred[t] += sred[t+s]; __syncthreads(); }
  if (t == 0) {
    float p1   = sred[0] + bsum;
    float mask = (ncnt > 0.f) ? 1.f : 0.f;
    float fp   = p1 / ((ncnt > 0.f) ? ncnt : 1.f);
    float tgt  = result[g];
    float ax = fabsf(fp);
    float sp_pos = fmaxf(fp, 0.f)  + log1pf(__expf(-ax)); // softplus(fp)
    float sp_neg = fmaxf(-fp, 0.f) + log1pf(__expf(-ax)); // softplus(-fp)
    float bce = tgt*sp_neg + (1.f - tgt)*sp_pos;
    atomicAdd(&red[0], bce*mask);
    atomicAdd(&red[1], mask);
    out[1 + g]       = sigf(fp)*mask;
    out[1 + GG + g]  = tgt*mask;
  }
}

__global__ void k_loss(const float* red, float* out) {
  if (threadIdx.x == 0) out[0] = red[0] / red[1];
}

// ---------------- launch ----------------
extern "C" void kernel_launch(void* const* d_in, const int* in_sizes, int n_in,
                              void* d_out, int out_size, void* d_ws, size_t ws_size,
                              hipStream_t stream) {
  (void)in_sizes; (void)n_in; (void)out_size; (void)ws_size;
  const int*   node_id     = (const int*)  d_in[2];
  const int*   edge        = (const int*)  d_in[3];
  const int*   edge_type   = (const int*)  d_in[4];
  const float* target_c    = (const float*)d_in[5];
  const float* result      = (const float*)d_in[6];
  const float* c_embed     = (const float*)d_in[7];
  const float* cur_result  = (const float*)d_in[8];
  const float* node_embedW = (const float*)d_in[9];
  const float* edge_embedW = (const float*)d_in[10];
  const float* ggnn_W      = (const float*)d_in[11];
  const float* gru_Wih     = (const float*)d_in[12];
  const float* gru_Whh     = (const float*)d_in[13];
  const float* gru_bih     = (const float*)d_in[14];
  const float* gru_bhh     = (const float*)d_in[15];
  const float* gate_W      = (const float*)d_in[16];
  const float* gate_b      = (const float*)d_in[17];
  const float* lstm_Wih    = (const float*)d_in[18];
  const float* lstm_Whh    = (const float*)d_in[19];
  const float* lstm_bih    = (const float*)d_in[20];
  const float* lstm_bhh    = (const float*)d_in[21];
  const float* pred_W      = (const float*)d_in[22];
  const float* pred_b      = (const float*)d_in[23];
  float* out = (float*)d_out;

  char* base = (char*)d_ws;
  size_t off = 0;
  auto nxt = [&](size_t bytes) -> void* {
    void* p = base + off; off += (bytes + 255) & ~(size_t)255; return p;
  };
  float*   ws_x    = (float*)  nxt((size_t)MROWS*DD*4);   // 52.4 MB
  __bf16*  ws_xbf  = (__bf16*) nxt((size_t)MROWS*DD*2);   // 26.2 MB
  __bf16*  ws_mbf  = (__bf16*) nxt((size_t)MROWS*DD*2);   // 26.2 MB (msg, then aggr in-place)
  float*   ws_ew   = (float*)  nxt((size_t)MROWS*4);
  float*   ws_tab  = (float*)  nxt(64);
  __bf16*  ws_pg   = (__bf16*) nxt((size_t)4*DD*DD*2);    // packed ggnn_W
  __bf16*  ws_pwih = (__bf16*) nxt((size_t)DD*384*2);
  __bf16*  ws_pwhh = (__bf16*) nxt((size_t)DD*384*2);
  __bf16*  ws_plih = (__bf16*) nxt((size_t)LKP*4*HH*2);   // packed lstm Wih (K pad 256)
  __bf16*  ws_plhh = (__bf16*) nxt((size_t)HH*4*HH*2);    // packed lstm Whh
  float*   ws_code = (float*)  nxt((size_t)GG*DD*4);
  float*   ws_lstm = (float*)  nxt((size_t)GG*HH*4);
  float*   ws_red  = (float*)  nxt(64);

  // prep
  k_ew_table <<<1, 32, 0, stream>>>(edge_embedW, ws_tab);
  k_ew_gather<<<(MROWS+255)/256, 256, 0, stream>>>(edge_type, ws_tab, ws_ew);
  k_embed    <<<4096, 256, 0, stream>>>(node_id, node_embedW, ws_x, ws_xbf);
  for (int l = 0; l < 4; ++l)
    k_pack_b<<<64, 256, 0, stream>>>(ggnn_W + (size_t)l*DD*DD, ws_pg + (size_t)l*DD*DD,
                                     DD, DD, DD, 0);
  k_pack_b<<<192, 256, 0, stream>>>(gru_Wih,  ws_pwih, DD,  3*DD, DD,   1);
  k_pack_b<<<192, 256, 0, stream>>>(gru_Whh,  ws_pwhh, DD,  3*DD, DD,   1);
  k_pack_b<<<1024, 256, 0, stream>>>(lstm_Wih, ws_plih, LKP, 4*HH, FDIM, 1);
  k_pack_b<<<1024, 256, 0, stream>>>(lstm_Whh, ws_plhh, HH,  4*HH, HH,   1);

  // 4 GGNN layers
  for (int l = 0; l < 4; ++l) {
    k_gemm_msg <<<dim3(MROWS/64, 2), 256, 0, stream>>>(ws_xbf, ws_pg + (size_t)l*DD*DD, ws_mbf);
    k_scatter  <<<GG, 256, NN*64*sizeof(float), stream>>>(ws_mbf, ws_ew, edge);
    k_gru_fused<<<MROWS/16, 256, 0, stream>>>(ws_mbf, ws_xbf, ws_pwih, ws_pwhh,
                                              gru_bih, gru_bhh, ws_x, ws_xbf);
  }

  // pool + LSTM + predict
  k_attn_pool<<<GG, 256, 0, stream>>>(ws_x, gate_W, gate_b, ws_code);
  k_lstm     <<<1, 1024, 0, stream>>>(c_embed, ws_code, cur_result,
                                      ws_plih, ws_plhh, lstm_bih, lstm_bhh, ws_lstm);
  k_init_red <<<1, 32, 0, stream>>>(ws_red);
  k_predict  <<<GG, 256, 0, stream>>>(ws_lstm, pred_W, pred_b, target_c, result, out, ws_red);
  k_loss     <<<1, 32, 0, stream>>>(ws_red, out);
}
